// InterpreterWithRegistersAndKbit_76072460747240
// MI455X (gfx1250) — matrix operand order repaired and checked by reading, not compiled, not run on hardware
//
#include <hip/hip_runtime.h>
#include <hip/hip_bf16.h>

// ---------------- problem constants ----------------
#define B_TOT   256
#define LINES_N 32
#define CAT_N   5
#define NREG_N  64
#define DDIM    1024
#define LDIM    64
#define KDIM    16
#define BT      16      // batch tile per workgroup (== WMMA M)
#define NWAVES  16
#define NTHR    512

typedef __attribute__((ext_vector_type(16))) __bf16 v16bf;
typedef __attribute__((ext_vector_type(8)))  float  v8f;

// ---------------- WMMA layout helpers ----------------
// B matrix (K=32 x N=16 bf16), contiguous k order used here: lanes 0-15 hold K=0..15 (N=lane),
// lanes 16-31 hold K=16..31 (N=lane-16); 16 contiguous bf16 per lane.
__device__ __host__ inline int wpackOff(int k, int n) {
    int nt = n >> 4, kt = k >> 5, kk = k & 31, nn = n & 15;
    int lane = (kk < 16) ? nn : (16 + nn);
    int e    = kk & 15;
    return (((nt * 32 + kt) * 32 + lane) << 4) + e;
}

// A matrix (M=16 x K=32 bf16), contiguous k order used here (same as B):
// lanes 0-15: K = 0..15; lanes 16-31: K = 16..31
__device__ inline int apackOff(int m, int k) {
    int kt = k >> 5, kk = k & 31;
    int lane, e;
    if (kk < 16) { lane = m;      e = kk;      }
    else         { lane = 16 + m; e = kk - 16; }
    return ((kt * 32 + lane) << 4) + e;
}

__device__ inline float waveSum(float v) {
#pragma unroll
    for (int off = 16; off > 0; off >>= 1) v += __shfl_xor(v, off, 32);
    return v;
}

// ---------------- weight pre-pack: f32 row-major -> bf16 WMMA-B tiles --------
__global__ void pack_weights(const float* __restrict__ W, __bf16* __restrict__ Wp) {
    int idx = blockIdx.x * blockDim.x + threadIdx.x;    // 0 .. 1M-1
    if (idx >= DDIM * DDIM) return;
    int k = idx >> 10, n = idx & 1023;                  // W[k][n], k = input dim
    Wp[wpackOff(k, n)] = (__bf16)W[idx];
}

// ---------------- main interpreter kernel ----------------
// grid = 16 (batch tiles), block = 512 threads = 16 wave32s
__global__ void interp_kernel(
    const float* __restrict__ prob,   // (B,32,5)
    const float* __restrict__ kwrite, // (B,32,1,1024)
    const float* __restrict__ qread,  // (B,64,1024)
    const float* __restrict__ gate,   // (B,32,1)
    const float* __restrict__ ln1g, const float* __restrict__ ln1b,
    const float* __restrict__ b1,
    const float* __restrict__ ln2g, const float* __restrict__ ln2b,
    const float* __restrict__ b2,
    const float* __restrict__ br,
    const float* __restrict__ lnfg, const float* __restrict__ lnfb,
    const __bf16* __restrict__ W1p, const __bf16* __restrict__ W2p,
    const __bf16* __restrict__ Wrp,
    float* __restrict__ regs)         // (B,64,1024) state, updated in place
{
    extern __shared__ char smem[];
    float*  sUnion  = (float*)smem;                        // 128KB: op_s -> logits -> value-logits
    __bf16* sA      = (__bf16*)(smem + 131072);            // 64KB: packed bf16 A (2 streams) / f32 value buf
    float*  sScores = (float*)(smem + 131072 + 65536);     // [16][3][64] = 12KB
    float*  sProb   = sScores + BT * 3 * NREG_N;           // [16][5]
    float*  sGate   = sProb + BT * CAT_N;                  // [16]

    const int tid  = threadIdx.x;
    const int wid  = tid >> 5;
    const int lane = tid & 31;
    const int b0   = blockIdx.x * BT;

    for (int s = 0; s < LINES_N; ++s) {
        // ---- phase 0: stage probs / gate ----
        if (tid < BT * CAT_N) {
            int b = tid / CAT_N, c = tid % CAT_N;
            sProb[b * CAT_N + c] = prob[((size_t)(b0 + b) * LINES_N + s) * CAT_N + c];
        }
        if (tid < BT) sGate[tid] = gate[(size_t)(b0 + tid) * LINES_N + s];
        __syncthreads();

        // ---- phase 1: attention scores (3 queries x 64 regs per batch) ----
        for (int u = wid; u < BT * 3; u += NWAVES) {
            int b = u / 3, j = u % 3;
            const float* qv = (j < 2)
                ? qread  + ((size_t)(b0 + b) * LINES_N * 2 + s * 2 + j) * DDIM
                : kwrite + ((size_t)(b0 + b) * LINES_N + s) * DDIM;
            const float*  rbase = regs + (size_t)(b0 + b) * NREG_N * DDIM;
            const float4* q4    = (const float4*)qv;
            for (int n = lane; n < NREG_N; n += 32) {
                const float4* r4 = (const float4*)(rbase + (size_t)n * DDIM);
                float acc = 0.f;
                for (int i = 0; i < DDIM / 4; ++i) {
                    float4 a = r4[i], qq = q4[i];
                    acc += a.x * qq.x + a.y * qq.y + a.z * qq.z + a.w * qq.w;
                }
                sScores[u * NREG_N + n] = acc * 0.03125f;   // 1/sqrt(1024)
            }
        }
        __syncthreads();

        // ---- phase 1b: softmax over 64 registers, one unit per thread ----
        if (tid < BT * 3) {
            float* sc = sScores + tid * NREG_N;
            float mx = -1e30f;
            for (int n = 0; n < NREG_N; ++n) mx = fmaxf(mx, sc[n]);
            float sum = 0.f;
            for (int n = 0; n < NREG_N; ++n) { float e = __expf(sc[n] - mx); sc[n] = e; sum += e; }
            float inv = 1.f / sum;
            for (int n = 0; n < NREG_N; ++n) sc[n] *= inv;
        }
        __syncthreads();

        // ---- phase 2: op_s = attn @ regs  (32 units of (b,q)) ----
        for (int u = wid; u < BT * 2; u += NWAVES) {
            int b = u >> 1, q = u & 1;
            const float* at    = sScores + (b * 3 + q) * NREG_N;
            const float* rbase = regs + (size_t)(b0 + b) * NREG_N * DDIM;
            float4 acc[8];
#pragma unroll
            for (int c = 0; c < 8; ++c) acc[c] = make_float4(0.f, 0.f, 0.f, 0.f);
            for (int n = 0; n < NREG_N; ++n) {
                float a = at[n];
                const float4* r4 = (const float4*)(rbase + (size_t)n * DDIM);
#pragma unroll
                for (int c = 0; c < 8; ++c) {
                    float4 v = r4[lane + 32 * c];
                    acc[c].x += a * v.x; acc[c].y += a * v.y;
                    acc[c].z += a * v.z; acc[c].w += a * v.w;
                }
            }
            float4* o4 = (float4*)(sUnion + u * DDIM);
#pragma unroll
            for (int c = 0; c < 8; ++c) o4[lane + 32 * c] = acc[c];
        }
        __syncthreads();

        // ---- phase 3: LayerNorm + pack to bf16 WMMA-A tiles ----
        for (int u = wid; u < BT * 2; u += NWAVES) {
            int b = u >> 1, q = u & 1;
            float* v = sUnion + u * DDIM;
            float lsum = 0.f, lsq = 0.f;
#pragma unroll 4
            for (int i = 0; i < 32; ++i) {
                float x = v[lane * 32 + i];
                lsum += x; lsq += x * x;
            }
            lsum = waveSum(lsum); lsq = waveSum(lsq);
            float mean = lsum * (1.f / DDIM);
            float var  = lsq * (1.f / DDIM) - mean * mean;
            float rs   = rsqrtf(var + 1e-5f);
            const float* g  = q ? ln2g : ln1g;
            const float* bb = q ? ln2b : ln1b;
            __bf16* A = sA + q * (32 * 32 * 16);
            for (int i = 0; i < 32; ++i) {
                int d = lane * 32 + i;
                float xv = (v[d] - mean) * rs * g[d] + bb[d];
                A[apackOff(b, d)] = (__bf16)xv;
            }
        }
        __syncthreads();

        // ---- phase 4: WMMA  [16x1024] @ W{1,2}[1024x1024] -> logits ----
        for (int r = 0; r < 8; ++r) {
            int t  = wid * 8 + r;
            int q  = t >> 6;
            int nt = t & 63;
            const __bf16* Ab = sA + q * (32 * 32 * 16);
            const __bf16* Wp = q ? W2p : W1p;
            v8f acc = {};
            for (int kt = 0; kt < 32; ++kt) {
                v16bf a  = *(const v16bf*)(Ab + (((kt * 32 + lane)) << 4));
                v16bf bm = *(const v16bf*)(Wp + (((nt * 32 + kt) * 32 + lane) << 4));
                if (kt < 31)
                    __builtin_prefetch(Wp + (((nt * 32 + kt + 1) * 32 + lane) << 4), 0, 1);
                acc = __builtin_amdgcn_wmma_f32_16x16x32_bf16(
                          false, a, false, bm, (short)0, acc, false, false);
            }
            int N = lane & 15, mhi = (lane >> 4) * 8;
            const float* bias = q ? b2 : b1;
            float bcol = bias[nt * 16 + N];
#pragma unroll
            for (int rr = 0; rr < 8; ++rr) {
                int M = rr + mhi;
                sUnion[(q * 16 + M) * DDIM + nt * 16 + N] = acc[rr] + bcol;
            }
        }
        __syncthreads();

        // ---- phase 5: per-(b,l) softmax over K, mod-K add/sub mix, pack bf16 ----
        for (int u = tid; u < BT * LDIM; u += NTHR) {
            int b = u >> 6, l = u & 63;
            float x[KDIM], y[KDIM];
#pragma unroll
            for (int m = 0; m < KDIM; ++m) {
                x[m] = sUnion[(0 * 16 + b) * DDIM + l * 16 + m];
                y[m] = sUnion[(1 * 16 + b) * DDIM + l * 16 + m];
            }
            float mx = -1e30f, my = -1e30f;
#pragma unroll
            for (int m = 0; m < KDIM; ++m) { mx = fmaxf(mx, x[m]); my = fmaxf(my, y[m]); }
            float sx = 0.f, sy = 0.f;
#pragma unroll
            for (int m = 0; m < KDIM; ++m) {
                x[m] = __expf(x[m] - mx); sx += x[m];
                y[m] = __expf(y[m] - my); sy += y[m];
            }
            float ix = 1.f / sx, iy = 1.f / sy;
#pragma unroll
            for (int m = 0; m < KDIM; ++m) { x[m] *= ix; y[m] *= iy; }
            float p1 = sProb[b * CAT_N + 1], p2 = sProb[b * CAT_N + 2];
            float p3 = sProb[b * CAT_N + 3], p4 = sProb[b * CAT_N + 4];
#pragma unroll
            for (int m = 0; m < KDIM; ++m) {
                float add = 0.f, sub = 0.f;
#pragma unroll
                for (int i = 0; i < KDIM; ++i) {
                    add += x[i] * y[(m - i) & 15];
                    sub += x[i] * y[(i - m) & 15];
                }
                float mix = p1 * x[m] + p2 * y[m] + p3 * add + p4 * sub;
                sA[apackOff(b, l * 16 + m)] = (__bf16)mix;   // mix A, stream-0 region
            }
        }
        __syncthreads();

        // ---- phase 6: WMMA  mix[16x1024] @ Wr -> value logits ----
        for (int r = 0; r < 4; ++r) {
            int nt = wid * 4 + r;
            v8f acc = {};
            for (int kt = 0; kt < 32; ++kt) {
                v16bf a  = *(const v16bf*)(sA + ((kt * 32 + lane) << 4));
                v16bf bm = *(const v16bf*)(Wrp + (((nt * 32 + kt) * 32 + lane) << 4));
                if (kt < 31)
                    __builtin_prefetch(Wrp + (((nt * 32 + kt + 1) * 32 + lane) << 4), 0, 1);
                acc = __builtin_amdgcn_wmma_f32_16x16x32_bf16(
                          false, a, false, bm, (short)0, acc, false, false);
            }
            int N = lane & 15, mhi = (lane >> 4) * 8;
            float bcol = br[nt * 16 + N];
#pragma unroll
            for (int rr = 0; rr < 8; ++rr)
                sUnion[(rr + mhi) * DDIM + nt * 16 + N] = acc[rr] + bcol;
        }
        __syncthreads();

        // ---- phase 7: final LN + gated erase/write of register file ----
        {
            int b = wid;                 // one wave per batch element
            float* v = sUnion + b * DDIM;
            float lsum = 0.f, lsq = 0.f;
#pragma unroll 4
            for (int i = 0; i < 32; ++i) {
                float x = v[lane * 32 + i];
                lsum += x; lsq += x * x;
            }
            lsum = waveSum(lsum); lsq = waveSum(lsq);
            float mean = lsum * (1.f / DDIM);
            float var  = lsq * (1.f / DDIM) - mean * mean;
            float rs   = rsqrtf(var + 1e-5f);
            float* val = (float*)sA + b * DDIM;     // reuse A region as f32 value buffer
            for (int i = 0; i < 32; ++i) {
                int d = lane * 32 + i;
                val[d] = (v[d] - mean) * rs * lnfg[d] + lnfb[d];
            }
            float geff = sGate[b] * (1.f - sProb[b * CAT_N + 0]);
            float* rbase = regs + (size_t)(b0 + b) * NREG_N * DDIM;
            const float4* v4 = (const float4*)val;
            for (int n = 0; n < NREG_N; ++n) {
                float w   = sScores[(b * 3 + 2) * NREG_N + n] * geff;
                float omw = 1.f - w;
                float4* r4 = (float4*)(rbase + (size_t)n * DDIM);
#pragma unroll
                for (int c = 0; c < 8; ++c) {
                    int d4 = lane + 32 * c;
                    float4 rv = r4[d4];
                    float4 vv = v4[d4];
                    rv.x = rv.x * omw + w * vv.x;
                    rv.y = rv.y * omw + w * vv.y;
                    rv.z = rv.z * omw + w * vv.z;
                    rv.w = rv.w * omw + w * vv.w;
                    r4[d4] = rv;
                }
            }
        }
        __threadfence();   // make register-file writes visible before next step's reads
        __syncthreads();
    }
}

// ---------------- host launch ----------------
extern "C" void kernel_launch(void* const* d_in, const int* in_sizes, int n_in,
                              void* d_out, int out_size, void* d_ws, size_t ws_size,
                              hipStream_t stream) {
    const float* prob      = (const float*)d_in[0];
    const float* registers = (const float*)d_in[1];
    const float* kwrite    = (const float*)d_in[2];
    const float* qread     = (const float*)d_in[3];
    const float* gatep     = (const float*)d_in[4];
    const float* ln1g      = (const float*)d_in[5];
    const float* ln1b      = (const float*)d_in[6];
    const float* W1        = (const float*)d_in[7];
    const float* b1        = (const float*)d_in[8];
    const float* ln2g      = (const float*)d_in[9];
    const float* ln2b      = (const float*)d_in[10];
    const float* W2        = (const float*)d_in[11];
    const float* b2        = (const float*)d_in[12];
    const float* Wr        = (const float*)d_in[13];
    const float* br        = (const float*)d_in[14];
    const float* lnfg      = (const float*)d_in[15];
    const float* lnfb      = (const float*)d_in[16];
    float* out = (float*)d_out;

    __bf16* W1p = reinterpret_cast<__bf16*>(d_ws);
    __bf16* W2p = W1p + (size_t)DDIM * DDIM;
    __bf16* Wrp = W2p + (size_t)DDIM * DDIM;

    // init register state in-place in d_out
    hipMemcpyAsync(out, registers,
                   (size_t)B_TOT * NREG_N * DDIM * sizeof(float),
                   hipMemcpyDeviceToDevice, stream);

    // pre-pack weights into bf16 WMMA-B tile layout
    const int pk_blocks = (DDIM * DDIM + 255) / 256;
    pack_weights<<<pk_blocks, 256, 0, stream>>>(W1, W1p);
    pack_weights<<<pk_blocks, 256, 0, stream>>>(W2, W2p);
    pack_weights<<<pk_blocks, 256, 0, stream>>>(Wr, Wrp);

    // 128KB union + 64KB A + 16KB scores/probs
    const size_t smem_bytes = 131072 + 65536 + 16384;
    interp_kernel<<<B_TOT / BT, NTHR, smem_bytes, stream>>>(
        prob, kwrite, qread, gatep,
        ln1g, ln1b, b1, ln2g, ln2b, b2, br, lnfg, lnfb,
        W1p, W2p, Wrp, out);
}
